// LSTMModel_326417515295
// MI455X (gfx1250) — matrix operand order repaired and checked by reading, not compiled, not run on hardware
//
#include <hip/hip_runtime.h>

// ---------------------------------------------------------------------------
// CDNA5 (gfx1250) LSTM: bf16 WMMA everywhere, wave32, persistent recurrence.
// B=32, T=2048, D=H=256, gates=1024, 2 layers, dense head 256->256.
// xg pre-activations stored step-transposed [t][col][b] so the serialized
// recurrence ingests them as contiguous v8f (b128) loads per D-tile column.
// ---------------------------------------------------------------------------

typedef __bf16 bf16;
typedef __attribute__((ext_vector_type(8)))  bf16  bfx8;
typedef __attribute__((ext_vector_type(16))) bf16  bfx16;
typedef __attribute__((ext_vector_type(8)))  float v8f;

union ABFrag { bfx16 v; bfx8 h[2]; };

static constexpr int T = 2048;

__device__ __forceinline__ v8f wmma_bf16(bfx16 a, bfx16 b, v8f c) {
  // D(f32 16x16) = A(bf16 16x32) * B(bf16 32x16) + C
  return __builtin_amdgcn_wmma_f32_16x16x32_bf16(false, a, false, b,
                                                 (short)0, c, false, false);
}

// B-fragment from a transposed weight matrix Wt[col][k] (k contiguous).
// B 32x16 frag, contiguous k order (as A): lane n = lane%16, K = kc*32 + 16*(lane>>4) + j.
__device__ __forceinline__ bfx16 load_b_frag_t(const bf16* __restrict__ wt,
                                               int col, int lane, int kc) {
  ABFrag f;
  const bf16* p = wt + (size_t)col * 256 + kc * 32 + ((lane >> 4) << 4);
  f.h[0] = *(const bfx8*)(p);
  f.h[1] = *(const bfx8*)(p + 8);
  return f.v;
}

// A-fragment from row-major bf16 matrix (256 cols). A 16x32, contiguous k order (as B):
// lane row = lane%16; K = kc*32 + 16*(lane>>4) + j.
__device__ __forceinline__ bfx16 load_a_frag(const bf16* __restrict__ rowp,
                                             int lane, int kc) {
  ABFrag f;
  int k0 = kc * 32 + ((lane >> 4) << 4);
  f.h[0] = *(const bfx8*)(rowp + k0);
  f.h[1] = *(const bfx8*)(rowp + k0 + 8);
  return f.v;
}

// ------------------------- prep / layout kernels ---------------------------

// x[B=32][T=2048][D=256] f32  ->  xb[(t*32+b)][256] bf16
__global__ void convert_x_kernel(const float* __restrict__ x,
                                 bf16* __restrict__ xb) {
  int row = blockIdx.x;              // t*32 + b
  int t = row >> 5, b = row & 31;
  int d = threadIdx.x;
  xb[(size_t)row * 256 + d] = (bf16)x[((size_t)b * T + t) * 256 + d];
}

// W[256][1024] (gate-major cols) -> Wt[c'][256] bf16, transposed + permuted:
// c' = wgk*256 + gate*64 + u   maps from   c = gate*256 + wgk*64 + u
__global__ void permute_wt_kernel(const float* __restrict__ W,
                                  bf16* __restrict__ Wt) {
  int cp  = blockIdx.x;
  int wgk = cp >> 8, r = cp & 255, gate = r >> 6, u = r & 63;
  int c   = gate * 256 + wgk * 64 + u;
  int k   = threadIdx.x;
  Wt[(size_t)cp * 256 + k] = (bf16)W[(size_t)k * 1024 + c];
}

__global__ void permute_bias_kernel(const float* __restrict__ b,
                                    float* __restrict__ bp) {
  int cp  = blockIdx.x * 256 + threadIdx.x;
  int wgk = cp >> 8, r = cp & 255, gate = r >> 6, u = r & 63;
  bp[cp] = b[gate * 256 + wgk * 64 + u];
}

// Wd[256][256] f32 -> Wdt[o][k] bf16 (transposed)
__global__ void transpose_wd_kernel(const float* __restrict__ Wd,
                                    bf16* __restrict__ Wdt) {
  int o = blockIdx.x, k = threadIdx.x;
  Wdt[(size_t)o * 256 + k] = (bf16)Wd[(size_t)k * 256 + o];
}

// --------------------- xg = A(bf16 Mx256) @ Wt(bf16) -----------------------
// C f32 stored step-transposed: C[t][col(1024)][b(32)].
// Block = 256 thr (8 waves), block tile 64x128 (rows = 2 t-steps), wave 32x32.
__global__ void gemm_xg_kernel(const bf16* __restrict__ A,
                               const bf16* __restrict__ Wt,
                               float* __restrict__ C) {
  int tid = threadIdx.x, lane = tid & 31, w = tid >> 5;
  int lrow = lane & 15, lhi = lane >> 4;
  int m0 = blockIdx.x * 64 + (w >> 2) * 32;   // multiple of 32 -> single t
  int n0 = blockIdx.y * 128 + (w & 3) * 32;
  v8f acc[2][2] = {};
#pragma unroll
  for (int kc = 0; kc < 8; ++kc) {
    bfx16 afr[2], bfr[2];
#pragma unroll
    for (int mt = 0; mt < 2; ++mt)
      afr[mt] = load_a_frag(A + (size_t)(m0 + mt * 16 + lrow) * 256, lane, kc);
#pragma unroll
    for (int nt = 0; nt < 2; ++nt)
      bfr[nt] = load_b_frag_t(Wt, n0 + nt * 16 + lrow, lane, kc);
#pragma unroll
    for (int mt = 0; mt < 2; ++mt)
#pragma unroll
      for (int nt = 0; nt < 2; ++nt)
        acc[mt][nt] = wmma_bf16(afr[mt], bfr[nt], acc[mt][nt]);
  }
  int trow = m0 >> 5;                          // time step of this wave's rows
  float* ct = C + (size_t)trow * 1024 * 32;
#pragma unroll
  for (int mt = 0; mt < 2; ++mt)
#pragma unroll
    for (int nt = 0; nt < 2; ++nt) {
      int col = n0 + nt * 16 + lrow;
      // lane holds 8 consecutive batch rows -> one 32B vector store
      *(v8f*)(ct + (size_t)col * 32 + mt * 16 + lhi * 8) = acc[mt][nt];
    }
}

// ------------------------- persistent recurrence ---------------------------
// 4 workgroups x 256 threads. WG k owns hidden units [64k,64k+64) i.e. the
// 256 permuted gate columns [256k,256k+256). Wh slice lives in 128KB LDS in
// WMMA B-fragment order. c-state stays in VGPRs. h exchanged via hseq in L2
// with a monotonic global-atomic barrier per step.
__global__ void lstm_rec_kernel(const float* __restrict__ xg,   // [T][1024][32] permuted cols
                                const bf16*  __restrict__ wht,  // [1024][256] perm+transposed
                                const float* __restrict__ bp,   // [1024] permuted
                                bf16*        __restrict__ hseq, // [T][32][256]
                                int*         __restrict__ counter,
                                int counter_base) {
  extern __shared__ bf16 ldsB[];      // [16 n-tiles][8 kc][32 lanes][16] = 128KB
  int tid = threadIdx.x, lane = tid & 31, w = tid >> 5;
  int wg  = blockIdx.x;               // 0..3
  int lrow = lane & 15, lhi = lane >> 4;

  // Stage this WG's Wh slice into LDS, pre-swizzled to B-fragment order.
  for (int idx = tid; idx < 16 * 8 * 32; idx += 256) {
    int nt = idx >> 8, kc = (idx >> 5) & 7, ln = idx & 31;
    int col = wg * 256 + nt * 16 + (ln & 15);
    int k0  = kc * 32 + ((ln >> 4) << 4);
    const bf16* src = wht + (size_t)col * 256 + k0;
    bf16* dst = ldsB + (size_t)idx * 16;
    *(bfx8*)dst       = *(const bfx8*)src;
    *(bfx8*)(dst + 8) = *(const bfx8*)(src + 8);
  }
  __syncthreads();

  int m_sel  = w >> 2;                 // 0..1 : batch rows [16*m_sel, +16)
  int n_base = w & 3;                  // 0..3 : 16 hidden units within gate block
  v8f cstate = {};                     // this wave's 16x16 slice of c
  int colbase = wg * 256 + n_base * 16 + lrow;   // + gate*64
  int hucol   = wg * 64 + n_base * 16 + lrow;    // hidden-unit column
  int bsub    = m_sel * 16 + lhi * 8;            // first of this lane's 8 rows

  for (int t = 0; t < T; ++t) {
    const float* xgp = xg + (size_t)t * 1024 * 32;
    if (t + 1 < T)
      __builtin_prefetch(xgp + (size_t)1024 * 32 + (size_t)colbase * 32 + bsub,
                         0, 1);
    v8f z[4];
#pragma unroll
    for (int g = 0; g < 4; ++g) {
      int col = colbase + g * 64;
      v8f pre = *(const v8f*)(xgp + (size_t)col * 32 + bsub);
      z[g] = pre + bp[col];
    }

    if (t > 0) {
      const bf16* hprev = hseq + (size_t)(t - 1) * 32 * 256;
      const bf16* arow  = hprev + (size_t)(m_sel * 16 + lrow) * 256;
#pragma unroll
      for (int kc = 0; kc < 8; ++kc) {
        bfx16 a = load_a_frag(arow, lane, kc);
#pragma unroll
        for (int g = 0; g < 4; ++g) {
          int nt = g * 4 + n_base;
          const bf16* bp8 = ldsB + (((size_t)nt * 8 + kc) * 32 + lane) * 16;
          ABFrag bfr;
          bfr.h[0] = *(const bfx8*)bp8;
          bfr.h[1] = *(const bfx8*)(bp8 + 8);
          z[g] = wmma_bf16(a, bfr.v, z[g]);
        }
      }
    }

    // Gate math: purely lane-local thanks to gate-interleaved permutation.
    bf16* hout = hseq + (size_t)t * 32 * 256;
#pragma unroll
    for (int v = 0; v < 8; ++v) {
      float ig = 1.0f / (1.0f + __expf(-z[0][v]));
      float fg = 1.0f / (1.0f + __expf(-z[1][v]));
      float gg = tanhf(z[2][v]);
      float og = 1.0f / (1.0f + __expf(-z[3][v]));
      float c  = fg * cstate[v] + ig * gg;
      cstate[v] = c;
      float h  = og * tanhf(c);
      int row = m_sel * 16 + v + lhi * 8;
      hout[(size_t)row * 256 + hucol] = (bf16)h;
    }

    // 4-WG step barrier (monotonic counter; reset by memset each launch).
    __threadfence();
    __syncthreads();
    if (tid == 0)
      __hip_atomic_fetch_add(counter, 1, __ATOMIC_RELEASE,
                             __HIP_MEMORY_SCOPE_AGENT);
    int target = counter_base + 4 * (t + 1);
    while (__hip_atomic_load(counter, __ATOMIC_ACQUIRE,
                             __HIP_MEMORY_SCOPE_AGENT) < target)
      __builtin_amdgcn_s_sleep(1);
    __syncthreads();
  }
}

// ----------------------------- dense head ----------------------------------
// out[32][256] f32 = hlast(bf16 32x256) @ Wd + bd.  One block, 8 waves.
__global__ void dense_out_kernel(const bf16* __restrict__ hlast,
                                 const bf16* __restrict__ wdt,  // [o][k]
                                 const float* __restrict__ bd,
                                 float* __restrict__ out) {
  int tid = threadIdx.x, lane = tid & 31, w = tid >> 5;
  int lrow = lane & 15, lhi = lane >> 4;
  int m_sel = w >> 2, n_base = w & 3;
  v8f acc[4] = {};
#pragma unroll
  for (int kc = 0; kc < 8; ++kc) {
    bfx16 a = load_a_frag(hlast + (size_t)(m_sel * 16 + lrow) * 256, lane, kc);
#pragma unroll
    for (int q = 0; q < 4; ++q) {
      int col = (q * 4 + n_base) * 16 + lrow;
      acc[q] = wmma_bf16(a, load_b_frag_t(wdt, col, lane, kc), acc[q]);
    }
  }
#pragma unroll
  for (int q = 0; q < 4; ++q) {
    int col = (q * 4 + n_base) * 16 + lrow;
    float bias = bd[col];
#pragma unroll
    for (int v = 0; v < 8; ++v) {
      int row = m_sel * 16 + v + lhi * 8;
      out[(size_t)row * 256 + col] = acc[q][v] + bias;
    }
  }
}

// ---------------------------------------------------------------------------

extern "C" void kernel_launch(void* const* d_in, const int* in_sizes, int n_in,
                              void* d_out, int out_size, void* d_ws, size_t ws_size,
                              hipStream_t stream) {
  const float* x   = (const float*)d_in[0];
  const float* Wi0 = (const float*)d_in[1];
  const float* Wh0 = (const float*)d_in[2];
  const float* b0  = (const float*)d_in[3];
  const float* Wi1 = (const float*)d_in[4];
  const float* Wh1 = (const float*)d_in[5];
  const float* b1  = (const float*)d_in[6];
  const float* Wd  = (const float*)d_in[7];
  const float* bd  = (const float*)d_in[8];
  float* out = (float*)d_out;

  char* ws = (char*)d_ws;
  size_t off = 0;
  auto alloc = [&](size_t bytes) -> char* {
    char* p = ws + off;
    off += (bytes + 255) & ~(size_t)255;
    return p;
  };
  int*   counter = (int*)  alloc(256);
  bf16*  xb      = (bf16*) alloc((size_t)65536 * 256 * 2);
  bf16*  hseq    = (bf16*) alloc((size_t)65536 * 256 * 2);
  float* xgbuf   = (float*)alloc((size_t)65536 * 1024 * 4);
  bf16*  Wi0t    = (bf16*) alloc((size_t)1024 * 256 * 2);
  bf16*  Wh0t    = (bf16*) alloc((size_t)1024 * 256 * 2);
  bf16*  Wi1t    = (bf16*) alloc((size_t)1024 * 256 * 2);
  bf16*  Wh1t    = (bf16*) alloc((size_t)1024 * 256 * 2);
  bf16*  Wdt     = (bf16*) alloc((size_t)256 * 256 * 2);
  float* b0p     = (float*)alloc(1024 * 4);
  float* b1p     = (float*)alloc(1024 * 4);

  hipMemsetAsync(counter, 0, 256, stream);

  convert_x_kernel<<<65536, 256, 0, stream>>>(x, xb);
  permute_wt_kernel<<<1024, 256, 0, stream>>>(Wi0, Wi0t);
  permute_wt_kernel<<<1024, 256, 0, stream>>>(Wh0, Wh0t);
  permute_wt_kernel<<<1024, 256, 0, stream>>>(Wi1, Wi1t);
  permute_wt_kernel<<<1024, 256, 0, stream>>>(Wh1, Wh1t);
  permute_bias_kernel<<<4, 256, 0, stream>>>(b0, b0p);
  permute_bias_kernel<<<4, 256, 0, stream>>>(b1, b1p);
  transpose_wd_kernel<<<256, 256, 0, stream>>>(Wd, Wdt);

  dim3 ggrid(1024, 8);
  gemm_xg_kernel<<<ggrid, 256, 0, stream>>>(xb, Wi0t, xgbuf);
  lstm_rec_kernel<<<4, 256, 131072, stream>>>(xgbuf, Wh0t, b0p, hseq,
                                              counter, 0);
  gemm_xg_kernel<<<ggrid, 256, 0, stream>>>(hseq, Wi1t, xgbuf);
  lstm_rec_kernel<<<4, 256, 131072, stream>>>(xgbuf, Wh1t, b1p, hseq,
                                              counter, 4 * 2048);
  dense_out_kernel<<<1, 256, 0, stream>>>(hseq + (size_t)(T - 1) * 32 * 256,
                                          Wdt, bd, out);
}
